// GPTBlock_32384053412199
// MI455X (gfx1250) — matrix operand order repaired and checked by reading, not compiled, not run on hardware
//
#include <hip/hip_runtime.h>

// ---------------------------------------------------------------------------
// GPT block for MI455X (gfx1250): bf16 WMMA GEMMs (async-LDS double-buffered)
// + causal flash attention.  All matmuls -> v_wmma_f32_16x16x32_bf16.
// ---------------------------------------------------------------------------

typedef __bf16 bf16_t;
typedef bf16_t v16bf __attribute__((ext_vector_type(16)));
typedef bf16_t v8bf  __attribute__((ext_vector_type(8)));
typedef float  v8f   __attribute__((ext_vector_type(8)));

#define D_MODEL 1024
#define NHEADS  16
#define DK      64
#define D_FF    4096
#define BATCH   4
#define SEQ     2048
#define MROWS   (BATCH * SEQ)   // 8192

// GEMM tiling
#define BM 128
#define BN 128
#define BK 32
#define LDP 40   // LDS row pitch in bf16 elems (80B): 16B aligned, conflict-free

// float -> bf16, round-to-nearest-even
__device__ inline bf16_t to_bf16(float f) {
  unsigned u = __builtin_bit_cast(unsigned, f);
  unsigned r = (u + 0x7fffu + ((u >> 16) & 1u)) >> 16;
  unsigned short h = (unsigned short)r;
  return __builtin_bit_cast(bf16_t, h);
}

__device__ inline v8f wmma_bf16(v16bf a, v16bf b, v8f c) {
  return __builtin_amdgcn_wmma_f32_16x16x32_bf16(false, a, false, b,
                                                 (short)0, c, false, false);
}

// A-matrix 16x32 bf16 fragment: lane<16 -> row=lane, K 0..15;
// lane>=16 -> row=lane-16, K 16..31.  Two 16B loads.
__device__ inline v16bf load_a_frag(const bf16_t* __restrict__ base, long ld,
                                    int row0, int k0, int lane) {
  int r = row0 + (lane & 15);
  int k = k0 + ((lane >> 4) << 4);
  const bf16_t* p = base + (long)r * ld + k;
  union { v16bf v; v8bf h[2]; } u;
  u.h[0] = *(const v8bf*)p;
  u.h[1] = *(const v8bf*)(p + 8);
  return u.v;
}

// B-matrix 32x16 bf16 fragment: lane<16 -> col=lane, K 0..15; lane>=16 ->
// col=lane-16, K 16..31.  B[k][n] = W[n][k] (y = x @ W^T): one 32B read
// from row n of W (or of the staged LDS tile).
__device__ inline v16bf load_b_frag(const bf16_t* __restrict__ W, long ld,
                                    int n0, int k0, int lane) {
  int n = n0 + (lane & 15);
  int k = k0 + ((lane >> 4) << 4);
  return *(const v16bf*)(W + (long)n * ld + k);
}

// Async copy of a 128x32 bf16 tile (row stride ldg elems) into LDS with pitch
// LDP.  256 threads x 2 chunks of 16B = 512 chunks.  ASYNCcnt += 2 per thread.
__device__ inline void async_copy_tile(const bf16_t* __restrict__ gbase,
                                       long ldg, bf16_t* lbuf, int tid) {
#pragma unroll
  for (int i = 0; i < 2; ++i) {
    int c = tid * 2 + i;
    int row = c >> 2;
    int col = (c & 3) * 8;                       // 8 bf16 = 16B chunk
    unsigned long long ga =
        (unsigned long long)(uintptr_t)(gbase + (long)row * ldg + col);
    unsigned lds_off = (unsigned)(uintptr_t)(lbuf + row * LDP + col);
    asm volatile("global_load_async_to_lds_b128 %0, %1, off"
                 :: "v"(lds_off), "v"(ga) : "memory");
  }
}

// ---------------------------------------------------------------------------
// f32 -> bf16 bulk convert
// ---------------------------------------------------------------------------
__global__ __launch_bounds__(256) void cvt_bf16_kernel(
    const float* __restrict__ in, bf16_t* __restrict__ out, long n) {
  long i = (long)blockIdx.x * blockDim.x + threadIdx.x;
  if (i < n) out[i] = to_bf16(in[i]);
}

// ---------------------------------------------------------------------------
// GEMM: C[M,N] = A[M,K](bf16) @ W[N,K]^T(bf16) + bias, fused epilogues.
// Block = 256 threads (8 waves), 128x128 tile; wave = 32x64 (2x4 WMMA).
// A/B K-slabs staged global->LDS with async b128 copies, double buffered.
// ---------------------------------------------------------------------------
enum { EPI_F32 = 0, EPI_QK_SPLIT = 1, EPI_V_SPLIT_T = 2, EPI_GELU_BF16 = 3 };

template <int EPI>
__global__ __launch_bounds__(256) void gemm_bf16_kernel(
    const bf16_t* __restrict__ A, const bf16_t* __restrict__ W,
    const float* __restrict__ bias, void* __restrict__ out,
    int M, int N, int K) {
  (void)M;
  const int tid  = threadIdx.x;
  const int lane = tid & 31;
  const int wave = tid >> 5;
  const int wm = wave & 3;          // 4 waves along M
  const int wn = wave >> 2;         // 2 waves along N
  const int m0 = blockIdx.y * BM + wm * 32;
  const int n0 = blockIdx.x * BN + wn * 64;

  __shared__ __align__(16) bf16_t lds[2][2][BM * LDP];  // [buf][A|B] 40KB

  const bf16_t* Ab = A + (long)(blockIdx.y * BM) * K;
  const bf16_t* Wb = W + (long)(blockIdx.x * BN) * K;

  v8f acc[2][4] = {};

  // preload stage 0
  async_copy_tile(Ab, K, &lds[0][0][0], tid);
  async_copy_tile(Wb, K, &lds[0][1][0], tid);

  int cur = 0;
  for (int kb = 0; kb < K; kb += BK) {
    if (kb + BK < K) {
      async_copy_tile(Ab + kb + BK, K, &lds[cur ^ 1][0][0], tid);
      async_copy_tile(Wb + kb + BK, K, &lds[cur ^ 1][1][0], tid);
      // 4 older copies must land; 4 newer may stay in flight (in-order done)
      asm volatile("s_wait_asynccnt 4" ::: "memory");
    } else {
      asm volatile("s_wait_asynccnt 0" ::: "memory");
    }
    __syncthreads();

    const bf16_t* lA = &lds[cur][0][0];
    const bf16_t* lB = &lds[cur][1][0];
    v16bf a0 = load_a_frag(lA, LDP, wm * 32,      0, lane);
    v16bf a1 = load_a_frag(lA, LDP, wm * 32 + 16, 0, lane);
#pragma unroll
    for (int nt = 0; nt < 4; ++nt) {
      v16bf b = load_b_frag(lB, LDP, wn * 64 + nt * 16, 0, lane);
      acc[0][nt] = wmma_bf16(a0, b, acc[0][nt]);
      acc[1][nt] = wmma_bf16(a1, b, acc[1][nt]);
    }
    __syncthreads();   // protect buffer that next iteration's asyncs overwrite
    cur ^= 1;
  }

  // C/D layout: vgpr r, lane<16 -> (M=r, N=lane); lane>=16 -> (M=r+8, N=lane-16)
  const int roff = (lane >> 4) << 3;
  const int ncol = lane & 15;
#pragma unroll
  for (int mi = 0; mi < 2; ++mi)
#pragma unroll
    for (int ni = 0; ni < 4; ++ni)
#pragma unroll
      for (int r = 0; r < 8; ++r) {
        int m = m0 + mi * 16 + roff + r;
        int n = n0 + ni * 16 + ncol;
        float v = acc[mi][ni][r] + bias[n];
        if (EPI == EPI_F32) {
          ((float*)out)[(long)m * N + n] = v;
        } else if (EPI == EPI_QK_SPLIT) {        // -> [B, H, S, DK] bf16
          int b = m >> 11, s = m & (SEQ - 1);
          int h = n >> 6,  d = n & 63;
          ((bf16_t*)out)[(((long)(b * NHEADS + h)) * SEQ + s) * DK + d] = to_bf16(v);
        } else if (EPI == EPI_V_SPLIT_T) {       // -> [B, H, DK, S] bf16
          int b = m >> 11, s = m & (SEQ - 1);
          int h = n >> 6,  d = n & 63;
          ((bf16_t*)out)[(((long)(b * NHEADS + h)) * DK + d) * SEQ + s] = to_bf16(v);
        } else {                                 // exact GELU -> bf16
          float g = 0.5f * v * (1.0f + erff(v * 0.70710678118654752f));
          ((bf16_t*)out)[(long)m * N + n] = to_bf16(g);
        }
      }
}

// ---------------------------------------------------------------------------
// Flash attention (causal). One wave per 16 query rows; online softmax.
// Q,K: [B*H, S, DK] bf16.  Vt: [B*H, DK, S] bf16.  ctx: [B*S, D_MODEL] bf16.
// ---------------------------------------------------------------------------
__global__ __launch_bounds__(128) void flash_attn_kernel(
    const bf16_t* __restrict__ Q, const bf16_t* __restrict__ Kc,
    const bf16_t* __restrict__ Vt, bf16_t* __restrict__ ctx) {
  const int lane = threadIdx.x & 31;
  const int wave = threadIdx.x >> 5;
  const int bh = blockIdx.y;
  const int q0 = (blockIdx.x * 4 + wave) * 16;

  // P tile staging: C/D layout -> A layout via LDS (pitch 48 keeps 16B align)
  __shared__ __align__(16) bf16_t pbuf[4][16][48];

  const bf16_t* Qb = Q  + (long)bh * SEQ * DK;
  const bf16_t* Kb = Kc + (long)bh * SEQ * DK;
  const bf16_t* Vb = Vt + (long)bh * DK * SEQ;

  v16bf qf0 = load_a_frag(Qb, DK, q0, 0,  lane);
  v16bf qf1 = load_a_frag(Qb, DK, q0, 32, lane);

  v8f o[4] = {};
  float mrow[8], lrow[8];
#pragma unroll
  for (int r = 0; r < 8; ++r) { mrow[r] = -3.0e30f; lrow[r] = 0.0f; }

  const int roff = (lane >> 4) << 3;   // 0 or 8
  const int ncol = lane & 15;

  for (int kb = 0; kb < q0 + 16; kb += 32) {   // causal: only k-tiles <= q
    v8f sc[2];
#pragma unroll
    for (int c = 0; c < 2; ++c) {
      v16bf kf0 = load_b_frag(Kb, DK, kb + c * 16, 0,  lane);
      v16bf kf1 = load_b_frag(Kb, DK, kb + c * 16, 32, lane);
      v8f z = {};
      sc[c] = wmma_bf16(qf1, kf1, wmma_bf16(qf0, kf0, z));
    }
#pragma unroll
    for (int r = 0; r < 8; ++r) {
      int q = q0 + roff + r;
      float s0 = sc[0][r] * 0.125f;            // 1/sqrt(64)
      float s1 = sc[1][r] * 0.125f;
      if (kb + ncol > q)      s0 = -3.0e30f;   // causal mask
      if (kb + 16 + ncol > q) s1 = -3.0e30f;
      float tmax = fmaxf(s0, s1);
#pragma unroll
      for (int off = 1; off < 16; off <<= 1)   // xor 1..8 stays in half-wave
        tmax = fmaxf(tmax, __shfl_xor(tmax, off));
      float newm = fmaxf(mrow[r], tmax);
      float corr = expf(mrow[r] - newm);
      float p0 = expf(s0 - newm);
      float p1 = expf(s1 - newm);
      float psum = p0 + p1;
#pragma unroll
      for (int off = 1; off < 16; off <<= 1)
        psum += __shfl_xor(psum, off);
      lrow[r] = lrow[r] * corr + psum;
      mrow[r] = newm;
#pragma unroll
      for (int nt = 0; nt < 4; ++nt) o[nt][r] *= corr;
      pbuf[wave][roff + r][ncol]      = to_bf16(p0);
      pbuf[wave][roff + r][16 + ncol] = to_bf16(p1);
    }
    // per-wave DS ops are in-order; wait for stores, block compiler reordering
    asm volatile("s_wait_dscnt 0" ::: "memory");
    union { v16bf v; v8bf h[2]; } pu;
    int prow = lane & 15;
    pu.h[0] = *(const v8bf*)&pbuf[wave][prow][2 * roff];
    pu.h[1] = *(const v8bf*)&pbuf[wave][prow][2 * roff + 8];
    // O += P(16x32) @ V(32x64): V column d over k is contiguous in Vt
#pragma unroll
    for (int nt = 0; nt < 4; ++nt) {
      v16bf vf = *(const v16bf*)(Vb + (long)(nt * 16 + ncol) * SEQ + kb +
                                 ((lane >> 4) << 4));
      o[nt] = wmma_bf16(pu.v, vf, o[nt]);
    }
  }

  const int b = bh >> 4, h = bh & 15;
#pragma unroll
  for (int nt = 0; nt < 4; ++nt)
#pragma unroll
    for (int r = 0; r < 8; ++r) {
      int q = q0 + roff + r;
      float val = o[nt][r] / lrow[r];
      ctx[((long)b * SEQ + q) * D_MODEL + h * DK + nt * 16 + ncol] = to_bf16(val);
    }
}

// ---------------------------------------------------------------------------
// residual + LayerNorm: out = LN(a + b) * gamma + beta.  One row per block.
// ---------------------------------------------------------------------------
__global__ __launch_bounds__(256) void resid_ln_kernel(
    const float* __restrict__ a, const float* __restrict__ b,
    const float* __restrict__ gamma, const float* __restrict__ beta,
    float* __restrict__ outf, bf16_t* __restrict__ outb) {
  const long row = blockIdx.x;
  const float* ap = a + row * D_MODEL;
  const float* bp = b + row * D_MODEL;
  float v[4];
  float s = 0.0f;
#pragma unroll
  for (int i = 0; i < 4; ++i) {
    int idx = threadIdx.x + i * 256;
    v[i] = ap[idx] + bp[idx];
    s += v[i];
  }
  __shared__ float red0[8], red1[8];
  const int lane = threadIdx.x & 31, wave = threadIdx.x >> 5;
#pragma unroll
  for (int off = 1; off < 32; off <<= 1) s += __shfl_xor(s, off);
  if (lane == 0) red0[wave] = s;
  __syncthreads();
  float tot = 0.0f;
#pragma unroll
  for (int w = 0; w < 8; ++w) tot += red0[w];
  float mu = tot * (1.0f / D_MODEL);
  float d2 = 0.0f;
#pragma unroll
  for (int i = 0; i < 4; ++i) { float d = v[i] - mu; d2 += d * d; }
#pragma unroll
  for (int off = 1; off < 32; off <<= 1) d2 += __shfl_xor(d2, off);
  if (lane == 0) red1[wave] = d2;
  __syncthreads();
  float vtot = 0.0f;
#pragma unroll
  for (int w = 0; w < 8; ++w) vtot += red1[w];
  float inv = rsqrtf(vtot * (1.0f / D_MODEL) + 1e-5f);
#pragma unroll
  for (int i = 0; i < 4; ++i) {
    int idx = threadIdx.x + i * 256;
    float o = (v[i] - mu) * inv * gamma[idx] + beta[idx];
    outf[row * D_MODEL + idx] = o;
    if (outb) outb[row * D_MODEL + idx] = to_bf16(o);
  }
}

// ---------------------------------------------------------------------------
// launch
// ---------------------------------------------------------------------------
extern "C" void kernel_launch(void* const* d_in, const int* in_sizes, int n_in,
                              void* d_out, int out_size, void* d_ws, size_t ws_size,
                              hipStream_t stream) {
  (void)in_sizes; (void)n_in; (void)out_size; (void)ws_size;
  const float* x   = (const float*)d_in[0];
  // d_in[1] = causal mask (implemented analytically)
  const float* wq  = (const float*)d_in[2];  const float* bq  = (const float*)d_in[3];
  const float* wk  = (const float*)d_in[4];  const float* bk  = (const float*)d_in[5];
  const float* wv  = (const float*)d_in[6];  const float* bv  = (const float*)d_in[7];
  const float* wo  = (const float*)d_in[8];  const float* bo  = (const float*)d_in[9];
  const float* w1  = (const float*)d_in[10]; const float* b1  = (const float*)d_in[11];
  const float* w2  = (const float*)d_in[12]; const float* b2  = (const float*)d_in[13];
  const float* g1  = (const float*)d_in[14]; const float* be1 = (const float*)d_in[15];
  const float* g2  = (const float*)d_in[16]; const float* be2 = (const float*)d_in[17];
  float* out = (float*)d_out;

  char* ws = (char*)d_ws;
  size_t off = 0;
  auto carve = [&](size_t bytes) -> char* {
    char* p = ws + off;
    off = (off + bytes + 255) & ~(size_t)255;
    return p;
  };
  bf16_t* xb   = (bf16_t*)carve((size_t)MROWS * D_MODEL * 2);
  bf16_t* wqb  = (bf16_t*)carve((size_t)D_MODEL * D_MODEL * 2);
  bf16_t* wkb  = (bf16_t*)carve((size_t)D_MODEL * D_MODEL * 2);
  bf16_t* wvb  = (bf16_t*)carve((size_t)D_MODEL * D_MODEL * 2);
  bf16_t* wob  = (bf16_t*)carve((size_t)D_MODEL * D_MODEL * 2);
  bf16_t* w1b  = (bf16_t*)carve((size_t)D_FF * D_MODEL * 2);
  bf16_t* w2b  = (bf16_t*)carve((size_t)D_MODEL * D_FF * 2);
  bf16_t* Qb   = (bf16_t*)carve((size_t)MROWS * D_MODEL * 2);
  bf16_t* Kb   = (bf16_t*)carve((size_t)MROWS * D_MODEL * 2);
  bf16_t* Vtb  = (bf16_t*)carve((size_t)MROWS * D_MODEL * 2);
  bf16_t* ctxb = (bf16_t*)carve((size_t)MROWS * D_MODEL * 2);
  float*  attnf= (float*) carve((size_t)MROWS * D_MODEL * 4);
  float*  hf   = (float*) carve((size_t)MROWS * D_MODEL * 4);
  bf16_t* hb   = (bf16_t*)carve((size_t)MROWS * D_MODEL * 2);
  bf16_t* ffb  = (bf16_t*)carve((size_t)MROWS * D_FF * 2);
  float*  ff2f = (float*) carve((size_t)MROWS * D_MODEL * 4);

  auto cvt = [&](const float* src, bf16_t* dst, long n) {
    cvt_bf16_kernel<<<dim3((unsigned)((n + 255) / 256)), 256, 0, stream>>>(src, dst, n);
  };
  cvt(x,  xb,  (long)MROWS * D_MODEL);
  cvt(wq, wqb, (long)D_MODEL * D_MODEL);
  cvt(wk, wkb, (long)D_MODEL * D_MODEL);
  cvt(wv, wvb, (long)D_MODEL * D_MODEL);
  cvt(wo, wob, (long)D_MODEL * D_MODEL);
  cvt(w1, w1b, (long)D_FF * D_MODEL);
  cvt(w2, w2b, (long)D_MODEL * D_FF);

  // QKV projections (fused head-split; V fused-transposed for PV B-frags)
  gemm_bf16_kernel<EPI_QK_SPLIT><<<dim3(D_MODEL / BN, MROWS / BM), 256, 0, stream>>>(
      xb, wqb, bq, Qb, MROWS, D_MODEL, D_MODEL);
  gemm_bf16_kernel<EPI_QK_SPLIT><<<dim3(D_MODEL / BN, MROWS / BM), 256, 0, stream>>>(
      xb, wkb, bk, Kb, MROWS, D_MODEL, D_MODEL);
  gemm_bf16_kernel<EPI_V_SPLIT_T><<<dim3(D_MODEL / BN, MROWS / BM), 256, 0, stream>>>(
      xb, wvb, bv, Vtb, MROWS, D_MODEL, D_MODEL);

  // causal flash attention: grid (S/64 q-tiles, B*H heads), 4 waves/block
  flash_attn_kernel<<<dim3(SEQ / 64, BATCH * NHEADS), 128, 0, stream>>>(Qb, Kb, Vtb, ctxb);

  // output projection
  gemm_bf16_kernel<EPI_F32><<<dim3(D_MODEL / BN, MROWS / BM), 256, 0, stream>>>(
      ctxb, wob, bo, attnf, MROWS, D_MODEL, D_MODEL);

  // h = LN1(x + attn_out)  (f32 + bf16 copies)
  resid_ln_kernel<<<MROWS, 256, 0, stream>>>(x, attnf, g1, be1, hf, hb);

  // FFN
  gemm_bf16_kernel<EPI_GELU_BF16><<<dim3(D_FF / BN, MROWS / BM), 256, 0, stream>>>(
      hb, w1b, b1, ffb, MROWS, D_FF, D_MODEL);
  gemm_bf16_kernel<EPI_F32><<<dim3(D_MODEL / BN, MROWS / BM), 256, 0, stream>>>(
      ffb, w2b, b2, ff2f, MROWS, D_MODEL, D_FF);

  // out = LN2(h + ffn)
  resid_ln_kernel<<<MROWS, 256, 0, stream>>>(hf, ff2f, g2, be2, out, nullptr);
}
